// LocalModSirenDecoder_9869834846265
// MI455X (gfx1250) — compile-verified, matrix operand order checked by reading, not run on hardware
//
#include <hip/hip_runtime.h>

typedef __attribute__((ext_vector_type(16))) _Float16 v16h;
typedef __attribute__((ext_vector_type(8)))  _Float16 v8h;
typedef __attribute__((ext_vector_type(8)))  float    v8f;

#define GG 32
#define GRID3 32768            // 32^3
#define LATC 32
#define HIDC 64
#define PTOT 400000            // B*N = 4*100000
#define NPB  100000
#define NW   4                 // waves per block
#define NBLK 625               // persistent blocks
#define TPW  10                // tiles per wave: 625*4*10*16 = 400000
#define NFRAG 44               // 4 (mw0) + 12 (mw1) + 12 (mw2) + 8 (sw1) + 8 (sw2)
#define FRAG_HALFS (NFRAG*512) // 32 lanes * 16 halfs per fragment

// Per-wave LDS scratch is ordered by the in-order LDS pipeline (same-wave DS
// ops execute in order on CDNA5); we only need to stop the *compiler* from
// reordering shared-memory accesses. Zero-instruction fence:
#define WAVE_LDS_FENCE() asm volatile("" ::: "memory")

// Single-instruction ReLU: v_med3_f32(x, 0, FLT_MAX) clamps to [0, FLT_MAX].
// NOTE: must use a FINITE upper bound -- fmed3(x,0,+inf) gets instcombined to
// maxnum(x,0) which costs an extra canonicalize instruction.
__device__ __forceinline__ float relu1(float x){
  return __builtin_amdgcn_fmed3f(x, 0.0f, 3.402823466e+38f);
}

__device__ __forceinline__ v8f splat8(float x){ v8f v={x,x,x,x,x,x,x,x}; return v; }
__device__ __forceinline__ v16h cat16(v8h a, v8h b){
  return __builtin_shufflevector(a,b,0,1,2,3,4,5,6,7,8,9,10,11,12,13,14,15);
}
__device__ __forceinline__ v8f wmma_f16(v16h a, v16h b, v8f c){
  return __builtin_amdgcn_wmma_f32_16x16x32_f16(false,a,false,b,(short)0,c,false,false);
}
// Latent A-fragment gather (16x32 f16): lane<16 -> K{0..7,16..23}, lane>=16 -> K{8..15,24..31}
__device__ __forceinline__ v16h gather_lv(const float* __restrict__ lvp, int hiH){
  v16h a;
  #pragma unroll
  for (int h=0; h<16; ++h){
    int ch = ((h>>3)<<4) + (hiH<<3) + (h&7);
    a[h] = (_Float16)lvp[(size_t)ch*GRID3];
  }
  return a;
}

// ---------------------------------------------------------------------------
// Setup: convert f32 weights to f16 CDNA5 WMMA B-fragments in d_ws.
// B (32x16 f16) lane layout (mirror of A layout, ISA 7.12.2):
//   lane n (0..15):  column n, halfs = K{0..7, 16..23}
//   lane n+16:       column n, halfs = K{8..15, 24..31}
// Fragment order per weight: f = kc*4 + nt  (4 N-tiles of 16 columns).
// ---------------------------------------------------------------------------
__global__ void swizzle_weights(const float* __restrict__ mw0,
                                const float* __restrict__ mw1,
                                const float* __restrict__ mw2,
                                const float* __restrict__ sw1,
                                const float* __restrict__ sw2,
                                _Float16* __restrict__ frags)
{
  int t = blockIdx.x*256 + threadIdx.x;
  if (t >= FRAG_HALFS) return;
  int frag = t >> 9;       // /512
  int r    = t & 511;
  int l    = r >> 4;       // lane 0..31
  int h    = r & 15;       // half 0..15
  const float* W; int fbase;
  if      (frag < 4)  { W = mw0; fbase = frag;      }
  else if (frag < 16) { W = mw1; fbase = frag - 4;  }
  else if (frag < 28) { W = mw2; fbase = frag - 16; }
  else if (frag < 36) { W = sw1; fbase = frag - 28; }
  else                { W = sw2; fbase = frag - 36; }
  int kc = fbase >> 2, nt = fbase & 3;
  int col = nt*16 + (l & 15);
  int klocal = ((h>>3)<<4) + ((l>>4)<<3) + (h&7);
  int k = kc*32 + klocal;
  frags[t] = (_Float16)W[k*HIDC + col];
}

// ---------------------------------------------------------------------------
// Main kernel: persistent, 4 waves/block, 16-point tiles, 8 corners per tile.
// ---------------------------------------------------------------------------
struct __align__(32) SMem {
  _Float16 frags[FRAG_HALFS];            // 45056 B, pre-swizzled B fragments
  _Float16 xpose[NW][16][72];            // per-wave activation transpose (pad 72)
  float mb[3][64];                       // modulator biases
  float sw0[192];                        // sir_w0 (3x64) row-major f32
  float sb[3][64];                       // siren biases
  float lastw[64];
  float red[NW][16][16];                 // final reduction buffer
  float x3[NW][16][3];                   // per-row x_ijk
  float lastb;
};

#define BFRAG(f) (*(const v16h*)(sm.frags + (((f)*32 + lane) << 4)))
#define LOAD_A(dst0, dst1)                                            \
  { const _Float16* pp=&xp[cl][(hiH<<3)];                             \
    dst0 = cat16(*(const v8h*)pp, *(const v8h*)(pp+16));              \
    const _Float16* pq=&xp[cl][32+(hiH<<3)];                          \
    dst1 = cat16(*(const v8h*)pq, *(const v8h*)(pq+16)); }
#define STORE_XP(src)                                                 \
  _Pragma("unroll")                                                   \
  for (int nt=0; nt<4; ++nt)                                          \
    _Pragma("unroll")                                                 \
    for (int r=0;r<8;++r) xp[r+rofs][nt*16+cl] = (_Float16)src[nt][r];

__global__ void __launch_bounds__(128) siren_main(
    const float* __restrict__ x,   const float* __restrict__ lv,
    const float* __restrict__ mb0, const float* __restrict__ mb1, const float* __restrict__ mb2,
    const float* __restrict__ sw0, const float* __restrict__ sb0, const float* __restrict__ sb1,
    const float* __restrict__ sb2, const float* __restrict__ lastw, const float* __restrict__ lastb,
    const _Float16* __restrict__ wsfrags, float* __restrict__ out)
{
  __shared__ SMem sm;
  const int tid = threadIdx.x;

  { // cooperative weight load (frags are 16B-aligned in ws and LDS)
    const uint4* src = (const uint4*)wsfrags;
    uint4* dst = (uint4*)sm.frags;
    for (int i = tid; i < FRAG_HALFS/8; i += 128) dst[i] = src[i];
    for (int i = tid; i < 64; i += 128){
      sm.mb[0][i]=mb0[i]; sm.mb[1][i]=mb1[i]; sm.mb[2][i]=mb2[i];
      sm.sb[0][i]=sb0[i]; sm.sb[1][i]=sb1[i]; sm.sb[2][i]=sb2[i];
      sm.lastw[i]=lastw[i];
    }
    for (int i = tid; i < 192; i += 128) sm.sw0[i]=sw0[i];
    if (tid==0) sm.lastb = lastb[0];
  }
  __syncthreads();   // only cross-wave sync in the kernel

  const int w = tid >> 5, lane = tid & 31;
  const int cl = lane & 15, hiH = lane >> 4, rofs = hiH*8;
  _Float16 (*xp)[72] = sm.xpose[w];

  #pragma unroll 1
  for (int t = 0; t < TPW; ++t){
    const int tile = blockIdx.x*NW + w + t*(NBLK*NW);
    const int p = tile*16 + cl;          // both half-waves mirror row = lane&15
    const int bi = p / NPB;

    float xg[3], lerp3[3]; int lo3[3], hi3[3];
    #pragma unroll
    for (int d=0; d<3; ++d){
      float xv = x[p*3+d]*32.0f;
      float g0 = truncf(xv-0.5f);
      float g1 = truncf(xv+0.5f);
      lo3[d] = min(max((int)g0,0),GG-1);
      hi3[d] = min(max((int)g1,0),GG-1);
      lerp3[d] = xv - 0.5f - g0;
      xg[d] = xv;
    }
    const float* lvb = lv + (size_t)bi*LATC*GRID3;

    // prime the software pipeline: gather latent for corner 0
    v16h alv = gather_lv(lvb + ((lo3[0]*GG+lo3[1])*GG+lo3[2]), hiH);

    float acc = 0.0f;
    #pragma unroll 1
    for (int c8=0; c8<8; ++c8){
      const int s0=(c8>>2)&1, s1=(c8>>1)&1, s2=c8&1;
      const int ci = s0?hi3[0]:lo3[0];
      const int cj = s1?hi3[1]:lo3[1];
      const int ck = s2?hi3[2]:lo3[2];
      const float xk0 = xg[0]-(float)ci-0.5f;
      const float xk1 = xg[1]-(float)cj-0.5f;
      const float xk2 = xg[2]-(float)ck-0.5f;

      // issue next corner's latent gather now; consumed next iteration
      v16h alv_nx = alv;
      if (c8 < 7){
        const int n8=c8+1;
        const int ni = ((n8>>2)&1)?hi3[0]:lo3[0];
        const int nj = ((n8>>1)&1)?hi3[1]:lo3[1];
        const int nk = ( n8     &1)?hi3[2]:lo3[2];
        alv_nx = gather_lv(lvb + ((ni*GG+nj)*GG+nk), hiH);
      }

      v8f m0[4], m1[4], m2[4];
      // ---- modulator layer 0: relu(lv @ mw0 + b0)  (4 WMMAs)
      #pragma unroll
      for (int nt=0; nt<4; ++nt){
        v8f c = splat8(sm.mb[0][nt*16+cl]);
        c = wmma_f16(alv, BFRAG(nt), c);
        #pragma unroll
        for (int r=0;r<8;++r) c[r] = relu1(c[r]);
        m0[nt]=c;
      }
      // both half-waves write identical values to identical addresses: no guard
      sm.x3[w][cl][0]=xk0; sm.x3[w][cl][1]=xk1; sm.x3[w][cl][2]=xk2;
      WAVE_LDS_FENCE();
      STORE_XP(m0);
      WAVE_LDS_FENCE();
      v16h a0, a1; LOAD_A(a0, a1);

      // ---- modulator layer 1: relu([h,z] @ mw1 + b1)  (12 WMMAs, kc2 = latent)
      #pragma unroll
      for (int nt=0; nt<4; ++nt){
        v8f c = splat8(sm.mb[1][nt*16+cl]);
        c = wmma_f16(a0,  BFRAG(4+nt),  c);
        c = wmma_f16(a1,  BFRAG(8+nt),  c);
        c = wmma_f16(alv, BFRAG(12+nt), c);
        #pragma unroll
        for (int r=0;r<8;++r) c[r] = relu1(c[r]);
        m1[nt]=c;
      }
      WAVE_LDS_FENCE();
      STORE_XP(m1);
      WAVE_LDS_FENCE();
      LOAD_A(a0, a1);

      // ---- modulator layer 2 (12 WMMAs)
      #pragma unroll
      for (int nt=0; nt<4; ++nt){
        v8f c = splat8(sm.mb[2][nt*16+cl]);
        c = wmma_f16(a0,  BFRAG(16+nt), c);
        c = wmma_f16(a1,  BFRAG(20+nt), c);
        c = wmma_f16(alv, BFRAG(24+nt), c);
        #pragma unroll
        for (int r=0;r<8;++r) c[r] = relu1(c[r]);
        m2[nt]=c;
      }

      // ---- SIREN layer 0 (K=3, VALU): sin(30*(x@sw0+b)) * m0
      float xr0[8], xr1[8], xr2[8];
      #pragma unroll
      for (int r=0;r<8;++r){
        int row = r+rofs;
        xr0[r]=sm.x3[w][row][0]; xr1[r]=sm.x3[w][row][1]; xr2[r]=sm.x3[w][row][2];
      }
      v8f hc[4];
      #pragma unroll
      for (int nt=0; nt<4; ++nt){
        int col = nt*16+cl;
        float w0c=sm.sw0[col], w1c=sm.sw0[64+col], w2c=sm.sw0[128+col], bc=sm.sb[0][col];
        v8f c;
        #pragma unroll
        for (int r=0;r<8;++r){
          float pre = fmaf(xr0[r],w0c, fmaf(xr1[r],w1c, fmaf(xr2[r],w2c,bc)));
          c[r] = __sinf(30.0f*pre) * m0[nt][r];
        }
        hc[nt]=c;
      }
      WAVE_LDS_FENCE();
      STORE_XP(hc);
      WAVE_LDS_FENCE();
      LOAD_A(a0, a1);

      // ---- SIREN layer 1 (8 WMMAs): sin(h@sw1+b) * m1
      #pragma unroll
      for (int nt=0; nt<4; ++nt){
        v8f c = splat8(sm.sb[1][nt*16+cl]);
        c = wmma_f16(a0, BFRAG(28+nt), c);
        c = wmma_f16(a1, BFRAG(32+nt), c);
        #pragma unroll
        for (int r=0;r<8;++r) c[r] = __sinf(c[r]) * m1[nt][r];
        hc[nt]=c;
      }
      WAVE_LDS_FENCE();
      STORE_XP(hc);
      WAVE_LDS_FENCE();
      LOAD_A(a0, a1);

      // ---- SIREN layer 2 (8 WMMAs): sin(h@sw2+b) * m2
      #pragma unroll
      for (int nt=0; nt<4; ++nt){
        v8f c = splat8(sm.sb[2][nt*16+cl]);
        c = wmma_f16(a0, BFRAG(36+nt), c);
        c = wmma_f16(a1, BFRAG(40+nt), c);
        #pragma unroll
        for (int r=0;r<8;++r) c[r] = __sinf(c[r]) * m2[nt][r];
        hc[nt]=c;
      }

      // ---- head 64->1: lane-partial then per-wave LDS reduction
      #pragma unroll
      for (int r=0;r<8;++r){
        float s = 0.0f;
        #pragma unroll
        for (int nt=0; nt<4; ++nt) s = fmaf(hc[nt][r], sm.lastw[nt*16+cl], s);
        sm.red[w][r+rofs][cl] = s;
      }
      WAVE_LDS_FENCE();
      { // redundantly computed in all 32 lanes (both halves produce row `cl`)
        float s = sm.lastb;
        #pragma unroll
        for (int q=0;q<16;++q) s += sm.red[w][cl][q];
        float cw = (s0?lerp3[0]:1.0f-lerp3[0]) *
                   (s1?lerp3[1]:1.0f-lerp3[1]) *
                   (s2?lerp3[2]:1.0f-lerp3[2]);
        acc = fmaf(cw, s, acc);
      }
      WAVE_LDS_FENCE();
      alv = alv_nx;
    }
    if (lane < 16) out[p] = acc;
  }
}

extern "C" void kernel_launch(void* const* d_in, const int* in_sizes, int n_in,
                              void* d_out, int out_size, void* d_ws, size_t ws_size,
                              hipStream_t stream)
{
  const float* x    = (const float*)d_in[0];
  const float* lv   = (const float*)d_in[1];
  const float* mw0  = (const float*)d_in[2];
  const float* mb0  = (const float*)d_in[3];
  const float* mw1  = (const float*)d_in[4];
  const float* mb1  = (const float*)d_in[5];
  const float* mw2  = (const float*)d_in[6];
  const float* mb2  = (const float*)d_in[7];
  const float* sw0  = (const float*)d_in[8];
  const float* sb0  = (const float*)d_in[9];
  const float* sw1  = (const float*)d_in[10];
  const float* sb1  = (const float*)d_in[11];
  const float* sw2  = (const float*)d_in[12];
  const float* sb2  = (const float*)d_in[13];
  const float* lw   = (const float*)d_in[14];
  const float* lb   = (const float*)d_in[15];
  _Float16* frags = (_Float16*)d_ws;

  hipLaunchKernelGGL(swizzle_weights, dim3((FRAG_HALFS+255)/256), dim3(256), 0, stream,
                     mw0, mw1, mw2, sw1, sw2, frags);
  hipLaunchKernelGGL(siren_main, dim3(NBLK), dim3(128), 0, stream,
                     x, lv, mb0, mb1, mb2, sw0, sb0, sb1, sb2, lw, lb, frags,
                     (float*)d_out);
}
